// Head_26895085207596
// MI455X (gfx1250) — matrix operand order repaired and checked by reading, not compiled, not run on hardware
//
#include <hip/hip_runtime.h>
#include <hip/hip_bf16.h>

typedef __attribute__((ext_vector_type(16))) _Float16 v16h;
typedef __attribute__((ext_vector_type(8)))  _Float16 v8h;
typedef __attribute__((ext_vector_type(8)))  float    v8f;
typedef __attribute__((ext_vector_type(4)))  float    v4f;

#define BSZ   256
#define TSEQ  256
#define CDIM  384
#define HDIM  64
#define WT_STRIDE 392   // 384 + 8-half pad to break LDS bank conflicts
#define PSTRIDE   40    // 32 + 8-half pad for P staging tile

// ---------------------------------------------------------------------------
// Kernel 1: fused QKV projection.  y = blockIdx.y selects {Wq, Wk, Wv}.
// Each block: 128 rows of x, weight matrix staged f32->f16 transposed in LDS.
// Each wave: one 16-row tile, 4 N-tiles of 16 over H=64, K-loop 12 x 32.
// Q,K stored row-major f16 [B*T, H]; V stored transposed f16 [B, H, T].
// ---------------------------------------------------------------------------
__global__ __launch_bounds__(256) void qkv_proj_kernel(
    const float* __restrict__ x,
    const float* __restrict__ Wk, const float* __restrict__ Wq,
    const float* __restrict__ Wv,
    _Float16* __restrict__ qh, _Float16* __restrict__ kh,
    _Float16* __restrict__ vT)
{
    __shared__ _Float16 wt[HDIM * WT_STRIDE];   // W^T in f16, padded rows

    const int y = blockIdx.y;
    const float* __restrict__ W = (y == 0) ? Wq : (y == 1) ? Wk : Wv;

    // cooperative load + transpose + f32->f16 convert into LDS
    for (int idx = threadIdx.x; idx < CDIM * HDIM; idx += 256) {
        int kk = idx / HDIM;        // K index (row of W)
        int n  = idx % HDIM;        // N index (col of W), coalesced in global
        wt[n * WT_STRIDE + kk] = (_Float16)W[idx];
    }
    __syncthreads();

    const int lane  = threadIdx.x & 31;
    const int wv    = threadIdx.x >> 5;
    const int half  = lane >> 4;
    const int l15   = lane & 15;
    const int klo   = half ? 8 : 0;     // A-frag K base (per ISA A layout)
    const int klo2  = half ? 16 : 0;    // unused here; B-frag also uses klo
    const int rowTile = blockIdx.x * 128 + wv * 16;   // global row of x
    const int row     = rowTile + l15;                // this lane's A row

    v8f acc[4] = {};    // 4 N-tiles of 16 -> H = 64

    for (int kc = 0; kc < CDIM; kc += 32) {
        // ---- A fragment: x row slice, f32 -> f16 ----
        const float* xr = x + (size_t)row * CDIM + kc;
        v4f x0 = *(const v4f*)(xr + klo);
        v4f x1 = *(const v4f*)(xr + klo + 4);
        v4f x2 = *(const v4f*)(xr + klo + 16);
        v4f x3 = *(const v4f*)(xr + klo + 20);
        v16h a;
        #pragma unroll
        for (int i = 0; i < 4; ++i) {
            a[i]      = (_Float16)x0[i];
            a[4 + i]  = (_Float16)x1[i];
            a[8 + i]  = (_Float16)x2[i];
            a[12 + i] = (_Float16)x3[i];
        }
        // ---- 4 B fragments from LDS (contiguous thanks to transpose) ----
        #pragma unroll
        for (int nt = 0; nt < 4; ++nt) {
            const _Float16* bp = &wt[(nt * 16 + l15) * WT_STRIDE + kc + klo];
            v8h b0 = *(const v8h*)(bp);
            v8h b1 = *(const v8h*)(bp + 16);
            v16h bf;
            #pragma unroll
            for (int i = 0; i < 8; ++i) { bf[i] = b0[i]; bf[8 + i] = b1[i]; }
            acc[nt] = __builtin_amdgcn_wmma_f32_16x16x32_f16(
                false, a, false, bf, (short)0, acc[nt], false, false);
        }
    }

    if (y < 2) {
        // Q / K: row-major f16 [B*T, H]
        _Float16* __restrict__ dst = (y == 0) ? qh : kh;
        #pragma unroll
        for (int nt = 0; nt < 4; ++nt)
            #pragma unroll
            for (int r = 0; r < 8; ++r) {
                int rr = rowTile + r + half * 8;          // C/D row layout
                dst[(size_t)rr * HDIM + nt * 16 + l15] = (_Float16)acc[nt][r];
            }
    } else {
        // V: transposed f16 [B, H, T]; pack 8 rows -> one 128-bit store
        int bIdx = rowTile / TSEQ;
        int t0   = (rowTile % TSEQ) + half * 8;
        #pragma unroll
        for (int nt = 0; nt < 4; ++nt) {
            v8h pk;
            #pragma unroll
            for (int r = 0; r < 8; ++r) pk[r] = (_Float16)acc[nt][r];
            *(v8h*)(vT + ((size_t)(bIdx * HDIM + nt * 16 + l15)) * TSEQ + t0) = pk;
        }
    }
}

// ---------------------------------------------------------------------------
// Kernel 2: causal flash attention, one 16-query tile per wave.
// Online softmax state per lane: rows r=0..7 <-> matrix row r + 8*half.
// P routed through per-wave LDS tile to convert C/D layout -> A layout.
// ---------------------------------------------------------------------------
__global__ __launch_bounds__(256) void flash_attn_kernel(
    const _Float16* __restrict__ qh, const _Float16* __restrict__ kh,
    const _Float16* __restrict__ vT, float* __restrict__ out)
{
    __shared__ _Float16 plds[8][16 * PSTRIDE];

    const int lane = threadIdx.x & 31;
    const int wv   = threadIdx.x >> 5;
    const int half = lane >> 4;
    const int l15  = lane & 15;
    const int klo  = half ? 8 : 0;
    const int klo2 = half ? 16 : 0;
    const int bIdx = blockIdx.y;
    const int q0   = blockIdx.x * 128 + wv * 16;

    // Q A-fragments for H=64 (two K-chunks of 32), loaded once
    const _Float16* qrow = qh + ((size_t)(bIdx * TSEQ + q0 + l15)) * HDIM;
    v16h qa0, qa1;
    {
        v8h t0 = *(const v8h*)(qrow + klo2);
        v8h t1 = *(const v8h*)(qrow + klo2 + 8);
        v8h t2 = *(const v8h*)(qrow + 32 + klo2);
        v8h t3 = *(const v8h*)(qrow + 32 + klo2 + 8);
        #pragma unroll
        for (int i = 0; i < 8; ++i) {
            qa0[i] = t0[i]; qa0[8 + i] = t1[i];
            qa1[i] = t2[i]; qa1[8 + i] = t3[i];
        }
    }

    float m_run[8], l_run[8];
    #pragma unroll
    for (int r = 0; r < 8; ++r) { m_run[r] = -1e30f; l_run[r] = 0.0f; }
    v8f o[4] = {};

    _Float16* P = &plds[wv][0];

    const int nIter = (q0 + 47) >> 5;   // causal: keys 0 .. q0+15 in steps of 32
    for (int it = 0; it < nIter; ++it) {
        const int j = it * 32;

        // ---- K B-fragments: 2 key-subtiles x 2 h-chunks (contiguous rows) ----
        const _Float16* kbase = kh + ((size_t)(bIdx * TSEQ + j + l15)) * HDIM;
        v16h kb00 = *(const v16h*)(kbase + klo2);               // keys j..j+15,  h 0..31
        v16h kb01 = *(const v16h*)(kbase + 32 + klo2);          // keys j..j+15,  h 32..63
        v16h kb10 = *(const v16h*)(kbase + 16 * HDIM + klo2);   // keys j+16..31, h 0..31
        v16h kb11 = *(const v16h*)(kbase + 16 * HDIM + 32 + klo2);

        // ---- S = Q K^T : two 16x16 tiles, K-dim chained over H ----
        v8f s0 = {}, s1 = {};
        s0 = __builtin_amdgcn_wmma_f32_16x16x32_f16(false, qa0, false, kb00, (short)0, s0, false, false);
        s0 = __builtin_amdgcn_wmma_f32_16x16x32_f16(false, qa1, false, kb01, (short)0, s0, false, false);
        s1 = __builtin_amdgcn_wmma_f32_16x16x32_f16(false, qa0, false, kb10, (short)0, s1, false, false);
        s1 = __builtin_amdgcn_wmma_f32_16x16x32_f16(false, qa1, false, kb11, (short)0, s1, false, false);

        // ---- scale + causal mask + online softmax ----
        const float scale = 0.125f;                 // 1/sqrt(64)
        const int qrow_base = q0 + half * 8;
        #pragma unroll
        for (int r = 0; r < 8; ++r) {
            float a0 = s0[r] * scale;
            float a1 = s1[r] * scale;
            const int qq = qrow_base + r;
            if (j + l15 > qq)      a0 = -1e30f;
            if (j + 16 + l15 > qq) a1 = -1e30f;
            float mx = fmaxf(a0, a1);
            #pragma unroll
            for (int off = 8; off >= 1; off >>= 1)
                mx = fmaxf(mx, __shfl_xor(mx, off, 32));
            float mnew  = fmaxf(m_run[r], mx);
            float alpha = __expf(m_run[r] - mnew);
            m_run[r] = mnew;
            float p0 = __expf(a0 - mnew);
            float p1 = __expf(a1 - mnew);
            float rs = p0 + p1;
            #pragma unroll
            for (int off = 8; off >= 1; off >>= 1)
                rs += __shfl_xor(rs, off, 32);
            l_run[r] = l_run[r] * alpha + rs;
            o[0][r] *= alpha; o[1][r] *= alpha;
            o[2][r] *= alpha; o[3][r] *= alpha;
            // stage P (C/D layout) into LDS
            const int m_row = r + half * 8;
            P[m_row * PSTRIDE + l15]      = (_Float16)p0;
            P[m_row * PSTRIDE + 16 + l15] = (_Float16)p1;
        }

        // LDS ops are in-order within a wave; explicit wait for safety
        asm volatile("s_wait_dscnt 0" ::: "memory");

        // ---- reload P in A-fragment layout ----
        v16h pa;
        {
            const _Float16* pr = &P[l15 * PSTRIDE + klo2];
            v8h u0 = *(const v8h*)(pr);
            v8h u1 = *(const v8h*)(pr + 8);
            #pragma unroll
            for (int i = 0; i < 8; ++i) { pa[i] = u0[i]; pa[8 + i] = u1[i]; }
        }

        // ---- O += P V : B-fragments contiguous thanks to transposed V ----
        const _Float16* vb = vT + ((size_t)bIdx * HDIM) * TSEQ + j + klo2;
        #pragma unroll
        for (int nt = 0; nt < 4; ++nt) {
            v16h bf = *(const v16h*)(vb + (size_t)(nt * 16 + l15) * TSEQ);
            o[nt] = __builtin_amdgcn_wmma_f32_16x16x32_f16(
                false, pa, false, bf, (short)0, o[nt], false, false);
        }
    }

    // ---- epilogue: normalize and store f32 [B, T, H] ----
    #pragma unroll
    for (int nt = 0; nt < 4; ++nt)
        #pragma unroll
        for (int r = 0; r < 8; ++r) {
            const int t = q0 + r + half * 8;
            out[((size_t)(bIdx * TSEQ + t)) * HDIM + nt * 16 + l15] =
                o[nt][r] / l_run[r];
        }
}

// ---------------------------------------------------------------------------
extern "C" void kernel_launch(void* const* d_in, const int* in_sizes, int n_in,
                              void* d_out, int out_size, void* d_ws, size_t ws_size,
                              hipStream_t stream) {
    const float* x  = (const float*)d_in[0];
    const float* Wk = (const float*)d_in[1];
    const float* Wq = (const float*)d_in[2];
    const float* Wv = (const float*)d_in[3];
    float* out = (float*)d_out;

    const size_t elems = (size_t)BSZ * TSEQ * HDIM;   // 4M elements, 8MB f16 each
    _Float16* qh = (_Float16*)d_ws;
    _Float16* kh = qh + elems;
    _Float16* vT = kh + elems;

    dim3 g1(BSZ * TSEQ / 128, 3);       // 512 row-blocks x {Q,K,V}
    qkv_proj_kernel<<<g1, 256, 0, stream>>>(x, Wk, Wq, Wv, qh, kh, vT);

    dim3 g2(TSEQ / 128, BSZ);           // 2 query-blocks x 256 batches
    flash_attn_kernel<<<g2, 256, 0, stream>>>(qh, kh, vT, out);
}
